// PointPatchTransformer_37598143709541
// MI455X (gfx1250) — matrix operand order repaired and checked by reading, not compiled, not run on hardware
//
#include <hip/hip_runtime.h>
#include <hip/hip_bf16.h>
#include <math.h>

// ---------------------------------------------------------------------------
// PointPatchTransformer forward for MI455X (gfx1250), wave32 + WMMA.
// All dense algebra runs through v_wmma_f32_16x16x32_f16 (f16 in, f32 acc).
// Fragment loads are contiguous b128 chunks (A: 2x16B, B: 2x16B per lane).
// ---------------------------------------------------------------------------

#define BB     8
#define NPTS   10000
#define PP     512
#define NSAMP  64
#define RAD2   0.04f
#define SADIM  128
#define DIMD   512
#define DEPTHL 12
#define HEADSN 8
#define DHEAD  64
#define MLPD   2048
#define SEQ    513            // P + 1 (cls token)
#define SEQP   544            // padded to 17 chunks of 32 keys
#define MROWS  (BB*SEQ)       // 4104
#define LIFTK  160            // 131 padded to 5*32
#define BN_SC  0.9999950000374997f   // 1/sqrt(1+1e-5)

typedef __attribute__((ext_vector_type(16))) _Float16 v16h;
typedef __attribute__((ext_vector_type(8)))  _Float16 v8h;
typedef __attribute__((ext_vector_type(8)))  float    v8f;

#define WMMA_F16(af, bf, cc) \
  __builtin_amdgcn_wmma_f32_16x16x32_f16(false, (af), false, (bf), (short)0, (cc), false, false)

__device__ __forceinline__ v16h pack16(v8h lo, v8h hi) {
  return __builtin_shufflevector(lo, hi, 0,1,2,3,4,5,6,7,8,9,10,11,12,13,14,15);
}

// A-matrix (16x32 f16): lane holds row (lane&15); halves are one contiguous
// 16-element run: k in [hi8, hi8+16), hi8 = (lane>=16)*16 (matches B).
// -> two 16B loads.
__device__ __forceinline__ v16h load_afrag(const _Float16* rowk0, int lane) {
  int hi8 = (lane >> 4) << 4;
  v8h lo = *(const v8h*)(rowk0 + hi8);
  v8h hi = *(const v8h*)(rowk0 + hi8 + 8);
  return pack16(lo, hi);
}
// B-matrix (32x16 f16): lane holds col (lane&15); halves are one contiguous
// 16-element run: k in [base, base+16), base = (lane>=16)*16. -> two 16B loads
// (requires K to be the contiguous axis of the source).
__device__ __forceinline__ v16h load_bfrag(const _Float16* colk0, int lane) {
  int base = (lane >> 4) << 4;
  v8h lo = *(const v8h*)(colk0 + base);
  v8h hi = *(const v8h*)(colk0 + base + 8);
  return pack16(lo, hi);
}

// ---------------------------------------------------------------------------
// 1. Furthest-point sampling: one block per batch, mind[] resident in LDS.
// ---------------------------------------------------------------------------
__global__ __launch_bounds__(1024)
void k_fps(const float* __restrict__ xyz, int* __restrict__ fidx, float* __restrict__ nxyz)
{
  int b = blockIdx.x;
  int tid = threadIdx.x, lane = tid & 31, wv = tid >> 5;
  __shared__ float mind[NPTS];
  __shared__ float wbest[32];
  __shared__ int   wbidx[32];
  __shared__ float slast[3];
  const float* X = xyz + (size_t)b * 3 * NPTS;
  const float* Y = X + NPTS;
  const float* Z = X + 2 * NPTS;
  for (int n = tid; n < NPTS; n += 1024) mind[n] = 1e10f;
  if (tid == 0) {
    fidx[b * PP] = 0;
    slast[0] = X[0]; slast[1] = Y[0]; slast[2] = Z[0];
    nxyz[(size_t)(b * PP) * 3 + 0] = X[0];
    nxyz[(size_t)(b * PP) * 3 + 1] = Y[0];
    nxyz[(size_t)(b * PP) * 3 + 2] = Z[0];
  }
  __syncthreads();
  for (int it = 1; it < PP; ++it) {
    float lx = slast[0], ly = slast[1], lz = slast[2];
    float best = -1.0f; int bi = 0;
    for (int n = tid; n < NPTS; n += 1024) {
      float dx = X[n] - lx, dy = Y[n] - ly, dz = Z[n] - lz;
      float d = dx * dx + dy * dy + dz * dz;
      float mn = mind[n] < d ? mind[n] : d;
      mind[n] = mn;
      if (mn > best || (mn == best && n < bi)) { best = mn; bi = n; }
    }
    for (int o = 16; o > 0; o >>= 1) {           // wave32 argmax reduce
      float ov = __shfl_xor(best, o, 32);
      int   oi = __shfl_xor(bi,   o, 32);
      if (ov > best || (ov == best && oi < bi)) { best = ov; bi = oi; }
    }
    if (lane == 0) { wbest[wv] = best; wbidx[wv] = bi; }
    __syncthreads();
    if (tid == 0) {
      float gb = wbest[0]; int gi = wbidx[0];
      for (int w = 1; w < 32; ++w)
        if (wbest[w] > gb || (wbest[w] == gb && wbidx[w] < gi)) { gb = wbest[w]; gi = wbidx[w]; }
      fidx[b * PP + it] = gi;
      slast[0] = X[gi]; slast[1] = Y[gi]; slast[2] = Z[gi];
      nxyz[(size_t)(b * PP + it) * 3 + 0] = X[gi];
      nxyz[(size_t)(b * PP + it) * 3 + 1] = Y[gi];
      nxyz[(size_t)(b * PP + it) * 3 + 2] = Z[gi];
    }
    __syncthreads();
  }
}

// ---------------------------------------------------------------------------
// 2. Ball query: one wave per patch; ballot+prefix append in index order.
// ---------------------------------------------------------------------------
__global__ __launch_bounds__(256)
void k_ball(const float* __restrict__ xyz, const float* __restrict__ nxyz, int* __restrict__ gidx)
{
  int patch = blockIdx.x * 8 + (threadIdx.x >> 5);
  int lane = threadIdx.x & 31;
  if (patch >= BB * PP) return;
  int b = patch / PP;
  const float* X = xyz + (size_t)b * 3 * NPTS;
  float cx = nxyz[(size_t)patch * 3 + 0];
  float cy = nxyz[(size_t)patch * 3 + 1];
  float cz = nxyz[(size_t)patch * 3 + 2];
  int* out = gidx + (size_t)patch * NSAMP;
  int count = 0, first = 0; bool haveFirst = false;
  for (int n0 = 0; n0 < NPTS && count < NSAMP; n0 += 32) {
    int n = n0 + lane;
    bool ok = false;
    if (n < NPTS) {
      float dx = X[n] - cx, dy = X[NPTS + n] - cy, dz = X[2 * NPTS + n] - cz;
      ok = (dx * dx + dy * dy + dz * dz) < RAD2;
    }
    unsigned m = (unsigned)__ballot(ok);
    if (!haveFirst && m) { first = n0 + (__ffs((int)m) - 1); haveFirst = true; }
    int pre = __popc(m & ((1u << lane) - 1u));
    int pos = count + pre;
    if (ok && pos < NSAMP) out[pos] = n;
    count += __popc(m);
  }
  if (count > NSAMP) count = NSAMP;
  for (int s = count + lane; s < NSAMP; s += 32) out[s] = first;  // center always in radius
}

// ---------------------------------------------------------------------------
// 3. PointNet SA MLP 6->64->64->128 + max pool. One block (64 thr) per patch.
// ---------------------------------------------------------------------------
__global__ __launch_bounds__(64)
void k_sa(const float* __restrict__ xyz, const float* __restrict__ fin,
          const float* __restrict__ nxyz, const int* __restrict__ gidx,
          const float* __restrict__ w1, const float* __restrict__ b1,
          const float* __restrict__ g1, const float* __restrict__ be1,
          const float* __restrict__ w2, const float* __restrict__ b2,
          const float* __restrict__ g2, const float* __restrict__ be2,
          const float* __restrict__ w3, const float* __restrict__ b3,
          const float* __restrict__ g3, const float* __restrict__ be3,
          float* __restrict__ featOut)
{
  int patch = blockIdx.x;
  int b = patch / PP;
  int t = threadIdx.x;                 // point slot 0..63
  __shared__ float sw1[64 * 6], sp1[64 * 3];
  __shared__ float sw2[64 * 64], sp2[64 * 3];
  __shared__ float sp3[128 * 3];
  __shared__ float h2s[64][65];
  for (int i = t; i < 64 * 6; i += 64) sw1[i] = w1[i];
  sp1[t] = b1[t]; sp1[64 + t] = g1[t]; sp1[128 + t] = be1[t];
  for (int i = t; i < 64 * 64; i += 64) sw2[i] = w2[i];
  sp2[t] = b2[t]; sp2[64 + t] = g2[t]; sp2[128 + t] = be2[t];
  for (int i = t; i < 128; i += 64) { sp3[i] = b3[i]; sp3[128 + i] = g3[i]; sp3[256 + i] = be3[i]; }
  __syncthreads();
  int n = gidx[(size_t)patch * NSAMP + t];
  const float* X = xyz + (size_t)b * 3 * NPTS;
  const float* F = fin + (size_t)b * 3 * NPTS;
  float in0[6];
  in0[0] = X[n]            - nxyz[(size_t)patch * 3 + 0];
  in0[1] = X[NPTS + n]     - nxyz[(size_t)patch * 3 + 1];
  in0[2] = X[2 * NPTS + n] - nxyz[(size_t)patch * 3 + 2];
  in0[3] = F[n]; in0[4] = F[NPTS + n]; in0[5] = F[2 * NPTS + n];
  float h1[64];
  for (int o = 0; o < 64; ++o) {
    float a = sp1[o];
    #pragma unroll
    for (int k = 0; k < 6; ++k) a += in0[k] * sw1[o * 6 + k];
    a = a * (sp1[64 + o] * BN_SC) + sp1[128 + o];
    h1[o] = a > 0.f ? a : 0.f;
  }
  for (int o = 0; o < 64; ++o) {
    float a = sp2[o];
    for (int k = 0; k < 64; ++k) a += h1[k] * sw2[o * 64 + k];
    a = a * (sp2[64 + o] * BN_SC) + sp2[128 + o];
    h2s[t][o] = a > 0.f ? a : 0.f;
  }
  __syncthreads();
  for (int cc = 0; cc < 2; ++cc) {     // layer3 channel-parallel + max pool
    int c = t + cc * 64;
    float wr[64];
    for (int k = 0; k < 64; ++k) wr[k] = w3[c * 64 + k];
    float bc = sp3[c], gc = sp3[128 + c] * BN_SC, ec = sp3[256 + c];
    float mx = -1e30f;
    for (int r = 0; r < 64; ++r) {
      float a = bc;
      for (int k = 0; k < 64; ++k) a += h2s[r][k] * wr[k];
      a = a * gc + ec;
      a = a > 0.f ? a : 0.f;
      if (a > mx) mx = a;
    }
    featOut[(size_t)patch * SADIM + c] = mx;
  }
}

// ---------------------------------------------------------------------------
// 4. Lift input assembly: [new_xyz(3) | feat(128) | 0-pad] -> f16, K=160
// ---------------------------------------------------------------------------
__global__ void k_liftprep(const float* __restrict__ nxyz, const float* __restrict__ feat,
                           _Float16* __restrict__ out)
{
  int i = blockIdx.x * blockDim.x + threadIdx.x;
  if (i >= BB * PP * LIFTK) return;
  int r = i / LIFTK, c = i % LIFTK;
  float v = 0.f;
  if (c < 3) v = nxyz[(size_t)r * 3 + c];
  else if (c < 131) v = feat[(size_t)r * SADIM + (c - 3)];
  out[i] = (_Float16)v;
}

// ---------------------------------------------------------------------------
// 5. Generic WMMA GEMM: C(M,N) = A(MxK f16) * W(K,N f32->f16) [+bias][gelu].
// Block = 4 waves, 64x64 tile; weight chunk staged K-contiguous in LDS so
// B-fragments are two ds_load_b128 per lane.
// mode 0: write f16   mode 1: write f32   mode 2: f32 residual +=
// ---------------------------------------------------------------------------
__global__ __launch_bounds__(128)
void k_gemm(const _Float16* __restrict__ A, int lda,
            const float* __restrict__ W, int ldb, int btrans,
            const float* __restrict__ bias,
            _Float16* __restrict__ out16, float* __restrict__ out32,
            float* __restrict__ resid,
            int M, int N, int K, int Kreal, int mode, int do_gelu)
{
  // Bt2[col][k], k contiguous; row padded 32->40 halves (80B) for banking.
  __shared__ __attribute__((aligned(16))) _Float16 Bt2[64][40];
  int tid = threadIdx.x, lane = tid & 31, wv = tid >> 5;
  int bm = blockIdx.x * 64, bn = blockIdx.y * 64;
  int rowm = bm + wv * 16 + (lane & 15);
  int rowc = rowm < M ? rowm : (M - 1);          // clamp; stores are guarded
  v8f acc[4] = {};
  for (int k0 = 0; k0 < K; k0 += 32) {
    if (k0 + 32 < K)                              // CDNA5 global_prefetch_b8
      __builtin_prefetch(btrans ? (const void*)&W[(size_t)bn * ldb + (k0 + 32)]
                                : (const void*)&W[(size_t)(k0 + 32) * ldb + bn], 0, 1);
    __syncthreads();
    if (!btrans) {
      for (int e = tid; e < 512; e += 128) {      // 512 float4 = 32x64 tile
        int kk = e >> 4, c4 = (e & 15) * 4;
        int kg = k0 + kk;
        float4 v4 = make_float4(0.f, 0.f, 0.f, 0.f);
        if (kg < Kreal) v4 = *(const float4*)&W[(size_t)kg * ldb + bn + c4];
        Bt2[c4 + 0][kk] = (_Float16)v4.x;
        Bt2[c4 + 1][kk] = (_Float16)v4.y;
        Bt2[c4 + 2][kk] = (_Float16)v4.z;
        Bt2[c4 + 3][kk] = (_Float16)v4.w;
      }
    } else {
      for (int e = tid; e < 2048; e += 128) {
        int kk = e & 31, c = e >> 5;
        int kg = k0 + kk;
        float v = (kg < Kreal) ? W[(size_t)(bn + c) * ldb + kg] : 0.f;
        Bt2[c][kk] = (_Float16)v;
      }
    }
    __syncthreads();
    v16h af = load_afrag(A + (size_t)rowc * lda + k0, lane);
    int n = lane & 15;
    #pragma unroll
    for (int nt = 0; nt < 4; ++nt) {
      v16h bf = load_bfrag(&Bt2[nt * 16 + n][0], lane);
      acc[nt] = WMMA_F16(af, bf, acc[nt]);
    }
  }
  int n0 = lane & 15;
  int mofs = (lane >> 4) << 3;                    // C layout: lanes>=16 -> rows+8
  #pragma unroll
  for (int nt = 0; nt < 4; ++nt) {
    #pragma unroll
    for (int r = 0; r < 8; ++r) {
      int m = bm + wv * 16 + mofs + r;
      int nn = bn + nt * 16 + n0;
      if (m < M) {
        float v = acc[nt][r];
        if (bias) v += bias[nn];
        if (do_gelu) v = 0.5f * v * (1.f + erff(v * 0.70710678118654752f));  // exact GELU
        size_t o = (size_t)m * N + nn;
        if (mode == 0) out16[o] = (_Float16)v;
        else if (mode == 1) out32[o] = v;
        else resid[o] += v;
      }
    }
  }
}

// ---------------------------------------------------------------------------
// 6. LayerNorm over DIMD=512. liftMap: row r -> token row r + r/PP + 1 of x.
// ---------------------------------------------------------------------------
__global__ __launch_bounds__(128)
void k_ln(const float* __restrict__ in, const float* __restrict__ g, const float* __restrict__ be,
          _Float16* __restrict__ out16, float* __restrict__ out32, int liftMap)
{
  int row = blockIdx.x, tid = threadIdx.x;
  const float* xr = in + (size_t)row * DIMD;
  float v[4], s = 0.f, s2 = 0.f;
  #pragma unroll
  for (int u = 0; u < 4; ++u) { v[u] = xr[tid + u * 128]; s += v[u]; s2 += v[u] * v[u]; }
  __shared__ float red[8];
  for (int o = 16; o > 0; o >>= 1) { s += __shfl_xor(s, o, 32); s2 += __shfl_xor(s2, o, 32); }
  if ((tid & 31) == 0) { red[tid >> 5] = s; red[4 + (tid >> 5)] = s2; }
  __syncthreads();
  s  = red[0] + red[1] + red[2] + red[3];
  s2 = red[4] + red[5] + red[6] + red[7];
  float mean = s * (1.f / DIMD);
  float var  = s2 * (1.f / DIMD) - mean * mean;
  float rs = rsqrtf(var + 1e-5f);
  size_t orow = (size_t)row + (liftMap ? (size_t)(row / PP) + 1 : 0);
  #pragma unroll
  for (int u = 0; u < 4; ++u) {
    int c = tid + u * 128;
    float y = (v[u] - mean) * rs * g[c] + be[c];
    if (out16) out16[orow * DIMD + c] = (_Float16)y;
    else       out32[orow * DIMD + c] = y;
  }
}

// ---------------------------------------------------------------------------
// 7. Positional term: pea[l,b,h,j] = cpad(b,j) . w_pe[l,h].
// (row-constant cpad_i.w_pe and b_pe cancel under softmax -> dropped)
// ---------------------------------------------------------------------------
__global__ void k_pea(const float* __restrict__ nxyz, const float* __restrict__ wpe,
                      float* __restrict__ pea)
{
  int i = blockIdx.x * blockDim.x + threadIdx.x;
  if (i >= DEPTHL * BB * HEADSN * SEQ) return;
  int j = i % SEQ;
  int h = (i / SEQ) % HEADSN;
  int lb = i / (SEQ * HEADSN);
  int b = lb % BB, l = lb / BB;
  float v = 0.f;
  if (j > 0) {
    const float* p = nxyz + (size_t)(b * PP + j - 1) * 3;
    const float* w = wpe + (size_t)(l * HEADSN + h) * 3;
    v = p[0] * w[0] + p[1] * w[1] + p[2] * w[2];
  }
  pea[i] = v;
}

// ---------------------------------------------------------------------------
// 8. Per-layer V transpose: vT[b,h,d][j] (keys padded to 544, zero-filled)
// so the P@V B-fragment is one contiguous 32B run per lane.
// ---------------------------------------------------------------------------
__global__ void k_vt(const _Float16* __restrict__ qkv, _Float16* __restrict__ vT)
{
  int i = blockIdx.x * blockDim.x + threadIdx.x;
  if (i >= BB * HEADSN * DHEAD * SEQP) return;
  int j = i % SEQP;
  int d = (i / SEQP) % DHEAD;
  int bh = i / (SEQP * DHEAD);
  int b = bh / HEADSN, h = bh % HEADSN;
  _Float16 v = (_Float16)0.0f;
  if (j < SEQ) v = qkv[(size_t)(b * SEQ + j) * 1536 + 1024 + h * 64 + d];
  vT[i] = v;
}

// ---------------------------------------------------------------------------
// 9. Attention: one block per (b,h,q-tile of 16). 16x544 score strip in LDS,
// shift-invariant softmax, then P@V — all matmuls through WMMA.
// ---------------------------------------------------------------------------
__global__ __launch_bounds__(128)
void k_attn(const _Float16* __restrict__ qkv, const _Float16* __restrict__ vT,
            const float* __restrict__ pea, _Float16* __restrict__ obuf, int layer)
{
  int bh = blockIdx.x;
  int b = bh / HEADSN, h = bh % HEADSN;
  int i0 = blockIdx.y * 16;
  int tid = threadIdx.x, lane = tid & 31, wv = tid >> 5;

  __shared__ __attribute__((aligned(16))) _Float16 qs[16][64];
  __shared__ float    sc[16][544];
  __shared__ __attribute__((aligned(16))) _Float16 pb[16][544];
  __shared__ float    rsum[16];

  const float* peaJ = pea + ((size_t)(layer * BB + b) * HEADSN + h) * SEQ;

  // load q tile: one v8h per thread (rows clamped; invalid rows masked later)
  {
    int m = tid >> 3, d8 = (tid & 7) * 8;
    int i = i0 + m; if (i > SEQ - 1) i = SEQ - 1;
    *(v8h*)&qs[m][d8] = *(const v8h*)(qkv + (size_t)(b * SEQ + i) * 1536 + h * 64 + d8);
  }
  __syncthreads();

  // scores: s = (q.k)/8 - pea_j ; each wave takes key tiles nt, nt+4, ...
  int mm = lane & 15;
  for (int nt = wv; nt < 34; nt += 4) {
    int j0 = nt * 16;
    int jn = j0 + (lane & 15); if (jn > SEQ - 1) jn = SEQ - 1;   // clamp, masked below
    const _Float16* krow = qkv + (size_t)(b * SEQ + jn) * 1536 + 512 + h * 64;
    v8f acc = {};
    #pragma unroll
    for (int ks = 0; ks < 2; ++ks) {
      int k0 = ks * 32;
      v16h af = load_afrag(&qs[mm][k0], lane);
      v16h bf = load_bfrag(krow + k0, lane);
      acc = WMMA_F16(af, bf, acc);
    }
    int n = lane & 15, mo = (lane >> 4) << 3;
    int j = j0 + n;
    float pj = (j < SEQ) ? peaJ[j] : 0.f;
    #pragma unroll
    for (int r = 0; r < 8; ++r) {
      float v = acc[r] * 0.125f - pj;            // scale = DH^-0.5 = 1/8
      sc[mo + r][j0 + n] = (j < SEQ) ? v : -1e30f;
    }
  }
  __syncthreads();

  // softmax per row (wave wv owns rows 4wv..4wv+3), unnormalized p -> f16
  for (int rr = 0; rr < 4; ++rr) {
    int r = wv * 4 + rr;
    int irow = i0 + r;
    if (irow >= SEQ) {
      if (lane == 0) rsum[r] = 1.f;
      for (int c = lane; c < 544; c += 32) pb[r][c] = (_Float16)0.0f;
      continue;
    }
    float mx = -1e30f;
    for (int c = lane; c < 544; c += 32) mx = fmaxf(mx, sc[r][c]);
    for (int o = 16; o > 0; o >>= 1) mx = fmaxf(mx, __shfl_xor(mx, o, 32));
    float sum = 0.f;
    for (int c = lane; c < 544; c += 32) {
      float e = __expf(sc[r][c] - mx);
      pb[r][c] = (_Float16)e;
      sum += e;
    }
    for (int o = 16; o > 0; o >>= 1) sum += __shfl_xor(sum, o, 32);
    if (lane == 0) rsum[r] = sum;
  }
  __syncthreads();

  // O = P @ V : wave wv owns v-dims [16wv, 16wv+16); K loop = 17 key chunks
  v8f oacc = {};
  int vcol = wv * 16 + (lane & 15);
  const _Float16* vrow = vT + ((size_t)(b * HEADSN + h) * DHEAD + vcol) * SEQP;
  for (int kc = 0; kc < 17; ++kc) {
    int k0 = kc * 32;
    v16h af = load_afrag(&pb[mm][k0], lane);
    v16h bf = load_bfrag(vrow + k0, lane);
    oacc = WMMA_F16(af, bf, oacc);
  }
  int mo = (lane >> 4) << 3;
  #pragma unroll
  for (int r = 0; r < 8; ++r) {
    int irow = i0 + mo + r;
    if (irow < SEQ)
      obuf[(size_t)(b * SEQ + irow) * 512 + h * 64 + vcol] =
          (_Float16)(oacc[r] / rsum[mo + r]);
  }
}

// ---------------------------------------------------------------------------
// small helpers
// ---------------------------------------------------------------------------
__global__ void k_setcls(const float* __restrict__ cls, float* __restrict__ x)
{
  int i = blockIdx.x * blockDim.x + threadIdx.x;
  if (i >= BB * DIMD) return;
  int b = i / DIMD, d = i % DIMD;
  x[(size_t)(b * SEQ) * DIMD + d] = cls[d];
}

__global__ void k_out(const float* __restrict__ x, float* __restrict__ out)
{
  int i = blockIdx.x * blockDim.x + threadIdx.x;
  if (i >= BB * DIMD) return;
  int b = i / DIMD, d = i % DIMD;
  out[i] = x[(size_t)(b * SEQ) * DIMD + d];
}

// ---------------------------------------------------------------------------
// host orchestration
// ---------------------------------------------------------------------------
extern "C" void kernel_launch(void* const* d_in, const int* in_sizes, int n_in,
                              void* d_out, int out_size, void* d_ws, size_t ws_size,
                              hipStream_t stream)
{
  (void)in_sizes; (void)n_in; (void)out_size; (void)ws_size;
  const float* xyz    = (const float*)d_in[0];
  const float* feats  = (const float*)d_in[1];
  const float* sa_w1  = (const float*)d_in[2];
  const float* sa_b1  = (const float*)d_in[3];
  const float* sa_g1  = (const float*)d_in[4];
  const float* sa_be1 = (const float*)d_in[5];
  const float* sa_w2  = (const float*)d_in[6];
  const float* sa_b2  = (const float*)d_in[7];
  const float* sa_g2  = (const float*)d_in[8];
  const float* sa_be2 = (const float*)d_in[9];
  const float* sa_w3  = (const float*)d_in[10];
  const float* sa_b3  = (const float*)d_in[11];
  const float* sa_g3  = (const float*)d_in[12];
  const float* sa_be3 = (const float*)d_in[13];
  const float* lift_w = (const float*)d_in[14];
  const float* lift_b = (const float*)d_in[15];
  const float* lift_g = (const float*)d_in[16];
  const float* lift_be= (const float*)d_in[17];
  const float* cls    = (const float*)d_in[18];
  const float* ln1_g  = (const float*)d_in[19];
  const float* ln1_b  = (const float*)d_in[20];
  const float* w_qkv  = (const float*)d_in[21];
  const float* w_o    = (const float*)d_in[22];
  const float* b_o    = (const float*)d_in[23];
  const float* w_pe   = (const float*)d_in[24];
  const float* b_pe   = (const float*)d_in[25];  (void)b_pe; // cancels in softmax
  const float* ln2_g  = (const float*)d_in[26];
  const float* ln2_b  = (const float*)d_in[27];
  const float* w_ff1  = (const float*)d_in[28];
  const float* b_ff1  = (const float*)d_in[29];
  const float* w_ff2  = (const float*)d_in[30];
  const float* b_ff2  = (const float*)d_in[31];

  char* ws = (char*)d_ws;
  size_t off = 0;
  auto alloc = [&](size_t bytes) -> void* {
    void* p = (void*)(ws + off);
    off = (off + bytes + 255) & ~(size_t)255;
    return p;
  };
  int*       fidx    = (int*)      alloc((size_t)BB * PP * 4);
  float*     nxyz    = (float*)    alloc((size_t)BB * PP * 3 * 4);
  int*       gidx    = (int*)      alloc((size_t)BB * PP * NSAMP * 4);
  float*     feat    = (float*)    alloc((size_t)BB * PP * SADIM * 4);
  _Float16*  liftin  = (_Float16*) alloc((size_t)BB * PP * LIFTK * 2);
  float*     liftout = (float*)    alloc((size_t)BB * PP * DIMD * 4);
  float*     x       = (float*)    alloc((size_t)MROWS * DIMD * 4);
  _Float16*  hn      = (_Float16*) alloc((size_t)MROWS * DIMD * 2);
  _Float16*  qkvb    = (_Float16*) alloc((size_t)MROWS * 1536 * 2);
  _Float16*  obuf    = (_Float16*) alloc((size_t)MROWS * DIMD * 2);
  _Float16*  ffb     = (_Float16*) alloc((size_t)MROWS * MLPD * 2);
  _Float16*  vT      = (_Float16*) alloc((size_t)BB * HEADSN * DHEAD * SEQP * 2);
  float*     pea     = (float*)    alloc((size_t)DEPTHL * BB * HEADSN * SEQ * 4);

  // ---- point-cloud front end ----
  k_fps <<<BB, 1024, 0, stream>>>(xyz, fidx, nxyz);
  k_ball<<<(BB * PP) / 8, 256, 0, stream>>>(xyz, nxyz, gidx);
  k_sa  <<<BB * PP, 64, 0, stream>>>(xyz, feats, nxyz, gidx,
                                     sa_w1, sa_b1, sa_g1, sa_be1,
                                     sa_w2, sa_b2, sa_g2, sa_be2,
                                     sa_w3, sa_b3, sa_g3, sa_be3, feat);

  // ---- lift + LN + cls ----
  k_liftprep<<<(BB * PP * LIFTK + 255) / 256, 256, 0, stream>>>(nxyz, feat, liftin);
  k_gemm<<<dim3((BB * PP) / 64, DIMD / 64), 128, 0, stream>>>(
      liftin, LIFTK, lift_w, 131, /*btrans*/1, lift_b,
      nullptr, liftout, nullptr, BB * PP, DIMD, LIFTK, 131, /*mode f32*/1, 0);
  k_ln<<<BB * PP, 128, 0, stream>>>(liftout, lift_g, lift_be, nullptr, x, /*liftMap*/1);
  k_setcls<<<(BB * DIMD + 255) / 256, 256, 0, stream>>>(cls, x);
  k_pea<<<(DEPTHL * BB * HEADSN * SEQ + 255) / 256, 256, 0, stream>>>(nxyz, w_pe, pea);

  // ---- transformer ----
  const int MT = (MROWS + 63) / 64;    // 65
  for (int l = 0; l < DEPTHL; ++l) {
    const float* l1g = ln1_g + (size_t)l * DIMD;
    const float* l1b = ln1_b + (size_t)l * DIMD;
    const float* wq  = w_qkv + (size_t)l * DIMD * 1536;
    const float* wo  = w_o   + (size_t)l * DIMD * DIMD;
    const float* bo  = b_o   + (size_t)l * DIMD;
    const float* l2g = ln2_g + (size_t)l * DIMD;
    const float* l2b = ln2_b + (size_t)l * DIMD;
    const float* wf1 = w_ff1 + (size_t)l * DIMD * MLPD;
    const float* bf1 = b_ff1 + (size_t)l * MLPD;
    const float* wf2 = w_ff2 + (size_t)l * MLPD * DIMD;
    const float* bf2 = b_ff2 + (size_t)l * DIMD;

    k_ln<<<MROWS, 128, 0, stream>>>(x, l1g, l1b, hn, nullptr, 0);
    k_gemm<<<dim3(MT, 1536 / 64), 128, 0, stream>>>(
        hn, DIMD, wq, 1536, 0, nullptr, qkvb, nullptr, nullptr,
        MROWS, 1536, DIMD, DIMD, 0, 0);
    k_vt<<<(BB * HEADSN * DHEAD * SEQP + 255) / 256, 256, 0, stream>>>(qkvb, vT);
    k_attn<<<dim3(BB * HEADSN, (SEQ + 15) / 16), 128, 0, stream>>>(qkvb, vT, pea, obuf, l);
    k_gemm<<<dim3(MT, DIMD / 64), 128, 0, stream>>>(
        obuf, DIMD, wo, DIMD, 0, bo, nullptr, nullptr, x,
        MROWS, DIMD, DIMD, DIMD, /*resid*/2, 0);
    k_ln<<<MROWS, 128, 0, stream>>>(x, l2g, l2b, hn, nullptr, 0);
    k_gemm<<<dim3(MT, MLPD / 64), 128, 0, stream>>>(
        hn, DIMD, wf1, MLPD, 0, bf1, ffb, nullptr, nullptr,
        MROWS, MLPD, DIMD, DIMD, 0, /*gelu*/1);
    k_gemm<<<dim3(MT, DIMD / 64), 128, 0, stream>>>(
        ffb, MLPD, wf2, DIMD, 0, bf2, nullptr, nullptr, x,
        MROWS, DIMD, MLPD, MLPD, /*resid*/2, 0);
  }

  k_out<<<(BB * DIMD + 255) / 256, 256, 0, stream>>>(x, (float*)d_out);
}
